// MultiLayerElmanRNN_17489106829417
// MI455X (gfx1250) — compile-verified, matrix operand order checked by reading, not run on hardware
//
#include <hip/hip_runtime.h>
#include <math.h>

// ---------------- problem constants ----------------
#define SEQ_LEN 512
#define BATCH   64
#define INDIM   512
#define HDIM    512
#define NLAYER  4

// ---------------- decomposition ----------------
#define HSLICE  128                       // hidden columns per workgroup
#define NSLICE  (HDIM / HSLICE)           // 4 workgroups per layer
#define RING    8                         // ring depth of hidden-state buffers

#define MT      (BATCH / 16)              // 4 m-tiles (batch)
#define KT      (HDIM / 32)               // 16 k-steps per 512-K GEMM
#define NT_SL   (HSLICE / 16)             // 8 n-tiles per slice

#define TILE_ELEMS  512                   // bf16 elements per fragment tile (32 lanes * 16)
#define TILE_BYTES  1024

// ---------------- LDS: swizzled bf16 weights, 2 mats x 16 kt x 8 nt tiles ----
#define WG_W_TILES  (2 * KT * NT_SL)                    // 256 tiles
#define LDS_BYTES   (WG_W_TILES * TILE_BYTES)           // 256 KB (< 320 KB WGP LDS)

// ---------------- workspace layout (flags + hidden-state ring only) --------
#define HBUF_OFF    8192                                 // flags: NLAYER*SEQ ints
#define ACT_TILES   (MT * KT)                            // 64 tiles = 64 KB per (layer,slot)
#define ACT_BYTES   (ACT_TILES * TILE_BYTES)
#define WS_ZERO_BYTES (HBUF_OFF + NLAYER * RING * ACT_BYTES)   // ~2.1 MB

typedef __bf16 bf16;
typedef bf16  v16bf __attribute__((ext_vector_type(16)));
typedef float v8f   __attribute__((ext_vector_type(8)));

// A-fragment (16x32 bf16, M on lanes, K on regs) element->K map:
//   lane < 16 : row M=lane,    K = {0..7, 16..23}
//   lane >=16 : row M=lane-16, K = {8..15, 24..31}
// Each lane's 16 bf16 stored contiguously (32 B). B mirrors A with N on lanes.

__device__ __forceinline__ void wait_ge(const int* f, int target) {
    if (threadIdx.x == 0) {
        while (__atomic_load_n(f, __ATOMIC_RELAXED) < target)
            __builtin_amdgcn_s_sleep(1);
    }
    __syncthreads();
}

__device__ __forceinline__ v16bf load_btile(const bf16* wtiles, int kt, int n_group,
                                            int nt, int lane) {
    return *(const v16bf*)(wtiles +
             (size_t)(kt * NT_SL + n_group * 4 + nt) * TILE_ELEMS + lane * 16);
}

// 4 WMMAs against LDS-resident swizzled weight tiles for one k-step
__device__ __forceinline__ void mm_step(v8f acc[4], v16bf a, const bf16* wtiles,
                                        int kt, int n_group, int lane) {
#pragma unroll
    for (int nt = 0; nt < 4; ++nt) {
        v16bf b = load_btile(wtiles, kt, n_group, nt, lane);
        acc[nt] = __builtin_amdgcn_wmma_f32_16x16x32_bf16(
            false, a, false, b, (short)0, acc[nt], false, false);
    }
}

// first k-step: C is the constant-zero vector -> lowers to inline SRC2=0
__device__ __forceinline__ void mm_first(v8f acc[4], v16bf a, const bf16* wtiles,
                                         int n_group, int lane) {
    const v8f z = {};
#pragma unroll
    for (int nt = 0; nt < 4; ++nt) {
        v16bf b = load_btile(wtiles, 0, n_group, nt, lane);
        acc[nt] = __builtin_amdgcn_wmma_f32_16x16x32_bf16(
            false, a, false, b, (short)0, z, false, false);
    }
}

__global__ void rnn_init_kernel(unsigned int* __restrict__ w, int nwords) {
    for (int i = blockIdx.x * blockDim.x + threadIdx.x; i < nwords;
         i += gridDim.x * blockDim.x)
        w[i] = 0u;
}

// Layer-0 A-fragment: gather + cvt from fp32 x rows
__device__ __forceinline__ v16bf load_a_x(const float* row, int kt, int hsel) {
    int K0 = kt * 32 + hsel * 8;
    v16bf a;
#pragma unroll
    for (int e = 0; e < 8; ++e) a[e]     = (bf16)row[K0 + e];
#pragma unroll
    for (int e = 0; e < 8; ++e) a[8 + e] = (bf16)row[K0 + 16 + e];
    return a;
}

__device__ __forceinline__ v16bf load_a_h(const bf16* abuf, int m_tile, int kt, int lane) {
    return *(const v16bf*)(abuf + (size_t)(m_tile * KT + kt) * TILE_ELEMS + lane * 16);
}

// The whole time loop, specialized on layer-0 (uniform per block) so the two
// A-source paths never share accumulator phis.
template <bool IS_L0>
__device__ __forceinline__ void rnn_time_loop(
        const float* __restrict__ x, float* __restrict__ out,
        int* __restrict__ flags, bf16* __restrict__ hbase,
        const bf16* wlds, const float bias[4],
        int layer, int slice, int m_tile, int n_group,
        int hsel, int ml, int lane, int tid) {

    const bf16* wmat_ih = wlds;
    const bf16* wmat_hh = wlds + (size_t)KT * NT_SL * TILE_ELEMS;

#pragma unroll 1
    for (int t = 0; t < SEQ_LEN; ++t) {
        // ---- dependency waits (wait graph strictly decreases 8t+l: deadlock-free)
        if (!IS_L0)     wait_ge(&flags[(layer - 1) * SEQ_LEN + t], NSLICE);
        if (t > 0)      wait_ge(&flags[layer * SEQ_LEN + (t - 1)], NSLICE);
        if (layer + 1 < NLAYER && t >= RING)
                        wait_ge(&flags[(layer + 1) * SEQ_LEN + (t - RING)], NSLICE);
        __threadfence();   // acquire: drop stale lines before re-reading ring slots

        v8f acc[4];

        // ---- GEMM 1: layer input (x for layer 0, else h_{l-1}(t)) x W_ih^T ----
        // first k-step writes acc with C = inline 0 (no accumulator init movs)
        if (IS_L0) {
            const float* row = x + ((size_t)t * BATCH + (m_tile * 16 + ml)) * INDIM;
            if (t + 1 < SEQ_LEN)   // warm L2 for next timestep's x rows (HBM-resident)
                __builtin_prefetch(row + (size_t)BATCH * INDIM, 0, 0);
            mm_first(acc, load_a_x(row, 0, hsel), wmat_ih, n_group, lane);
            for (int kt = 1; kt < KT; ++kt)
                mm_step(acc, load_a_x(row, kt, hsel), wmat_ih, kt, n_group, lane);
        } else {
            const bf16* abuf = hbase +
                ((size_t)(layer - 1) * RING + (t & (RING - 1))) * ACT_TILES * TILE_ELEMS;
            mm_first(acc, load_a_h(abuf, m_tile, 0, lane), wmat_ih, n_group, lane);
            for (int kt = 1; kt < KT; ++kt)
                mm_step(acc, load_a_h(abuf, m_tile, kt, lane), wmat_ih, kt, n_group, lane);
        }

        // ---- GEMM 2: recurrent h_l(t-1) x W_hh^T ----
        {
            const bf16* abuf = hbase +
                ((size_t)layer * RING + (((unsigned)(t - 1)) & (RING - 1))) * ACT_TILES * TILE_ELEMS;
            for (int kt = 0; kt < KT; ++kt)
                mm_step(acc, load_a_h(abuf, m_tile, kt, lane), wmat_hh, kt, n_group, lane);
        }

        // ---- epilogue: +bias, tanh, fp32 -> d_out, bf16 -> A-swizzled ring slot ----
        bf16* hout = hbase + ((size_t)layer * RING + (t & (RING - 1))) * ACT_TILES * TILE_ELEMS;
        float* obase = out + ((size_t)t * NLAYER + layer) * BATCH * HDIM;
#pragma unroll
        for (int nt = 0; nt < 4; ++nt) {
            int Ng   = slice * HSLICE + (n_group * 4 + nt) * 16 + ml;
            int kta  = Ng >> 5;                       // consumer k-tile
            int klo  = Ng & 31;
            int ha   = (klo >> 3) & 1;                // consumer fragment half
            int e    = ((klo & 16) ? 8 : 0) + (klo & 7);
#pragma unroll
            for (int r = 0; r < 8; ++r) {
                int M = m_tile * 16 + r + 8 * hsel;   // batch row of this C element
                float v = tanhf(acc[nt][r] + bias[nt]);
                obase[(size_t)M * HDIM + Ng] = v;
                hout[(size_t)(m_tile * KT + kta) * TILE_ELEMS +
                     ((M & 15) + 16 * ha) * 16 + e] = (bf16)v;
            }
        }

        // ---- publish: release stores, then one atomic per WG ----
        __threadfence();
        __syncthreads();
        if (tid == 0) atomicAdd(&flags[layer * SEQ_LEN + t], 1);
    }
}

__launch_bounds__(256, 1)
__global__ void rnn_persistent_kernel(const float* __restrict__ x,
                                      const float* __restrict__ W_ih,
                                      const float* __restrict__ W_hh,
                                      const float* __restrict__ b_ih,
                                      const float* __restrict__ b_hh,
                                      float* __restrict__ out,
                                      unsigned char* __restrict__ ws) {
    extern __shared__ unsigned char smem_raw[];
    bf16* wlds = (bf16*)smem_raw;                 // 256 KB swizzled bf16 weights

    const int wg    = blockIdx.x;        // 0..15
    const int layer = wg >> 2;
    const int slice = wg & (NSLICE - 1);
    const int tid   = threadIdx.x;
    const int lane  = tid & 31;
    const int wave  = tid >> 5;          // 0..7
    const int m_tile  = wave >> 1;       // 0..3 (batch tile)
    const int n_group = wave & 1;        // n-tiles n_group*4 .. +3
    const int hsel = lane >> 4;          // fragment half
    const int ml   = lane & 15;

    int*  flags = (int*)ws;
    bf16* hbase = (bf16*)(ws + HBUF_OFF);

    // ---------- one-time: convert this WG's weight slices to swizzled bf16
    // ---------- B-fragments, resident in LDS for the whole sequence ----------
    {
        const float* Wih_l = W_ih + (size_t)layer * HDIM * INDIM;
        const float* Whh_l = W_hh + (size_t)layer * HDIM * HDIM;
        for (int tile = wave; tile < WG_W_TILES; tile += 8) {
            int mat = tile >> 7;             // 0: W_ih, 1: W_hh
            int kt  = (tile >> 3) & (KT - 1);
            int nt  = tile & (NT_SL - 1);
            int Nrow = slice * HSLICE + nt * 16 + ml;   // output-neuron row of W
            int K0   = kt * 32 + hsel * 8;
            const float* src = (mat == 0 ? Wih_l : Whh_l) + (size_t)Nrow * 512 + K0;
            v16bf v;
#pragma unroll
            for (int e = 0; e < 8; ++e) v[e]     = (bf16)src[e];
#pragma unroll
            for (int e = 0; e < 8; ++e) v[8 + e] = (bf16)src[16 + e];
            *(v16bf*)(wlds + (size_t)tile * TILE_ELEMS + lane * 16) = v;
        }
    }

    // per-lane fused bias for each of this wave's 4 n-tiles
    float bias[4];
#pragma unroll
    for (int nt = 0; nt < 4; ++nt) {
        int Ng = slice * HSLICE + (n_group * 4 + nt) * 16 + ml;
        bias[nt] = b_ih[layer * HDIM + Ng] + b_hh[layer * HDIM + Ng];
    }

    __syncthreads();   // LDS weights visible to all waves in WG

    if (layer == 0)
        rnn_time_loop<true >(x, out, flags, hbase, wlds, bias,
                             layer, slice, m_tile, n_group, hsel, ml, lane, tid);
    else
        rnn_time_loop<false>(x, out, flags, hbase, wlds, bias,
                             layer, slice, m_tile, n_group, hsel, ml, lane, tid);
}

extern "C" void kernel_launch(void* const* d_in, const int* in_sizes, int n_in,
                              void* d_out, int out_size, void* d_ws, size_t ws_size,
                              hipStream_t stream) {
    (void)in_sizes; (void)n_in; (void)out_size; (void)ws_size;
    const float* x    = (const float*)d_in[0];
    const float* W_ih = (const float*)d_in[1];
    const float* W_hh = (const float*)d_in[2];
    const float* b_ih = (const float*)d_in[3];
    const float* b_hh = (const float*)d_in[4];
    float* out = (float*)d_out;
    unsigned char* ws = (unsigned char*)d_ws;

    // zero flags + all hidden-state ring buffers (h(-1) == 0), every call
    rnn_init_kernel<<<256, 256, 0, stream>>>((unsigned int*)ws, WS_ZERO_BYTES / 4);
    // persistent pipelined RNN: 16 co-resident workgroups (4 layers x 4 slices),
    // 256 KB dynamic LDS per WG holding the swizzled bf16 weight slices
    rnn_persistent_kernel<<<NLAYER * NSLICE, 256, LDS_BYTES, stream>>>(
        x, W_ih, W_hh, b_ih, b_hh, out, ws);
}
